// CNNMambaFusionModel_87179246174647
// MI455X (gfx1250) — matrix operand order repaired and checked by reading, not compiled, not run on hardware
//
#include <hip/hip_runtime.h>

// ---------------------------------------------------------------------------
// CNN-Mamba fusion for MI455X (gfx1250, wave32).
//   - GEMMs via v_wmma_f32_16x16x32_bf16, compile-time N (zero-padded weights)
//     -> no divergent guards, b128 fragment loads
//   - selective scan as a 3-pass chunked linear-recurrence scan (262k-way ||)
//   - activations stored bf16 (halve HBM bytes @ 23.3 TB/s), math in f32
// ---------------------------------------------------------------------------

#define BATCH 16
#define SEQ   4096
#define NTOK  (BATCH * SEQ)        // 65536 tokens
#define DM    128                  // d_model
#define DI    256                  // d_inner
#define DSN   16                   // d_state
#define DR    8                    // dt_rank
#define DBCW  40                   // dt_rank + 2*d_state (true width)
#define DBCP  48                   // padded to multiple of 16 for WMMA N
#define NCHK  64                   // scan chunks
#define CLEN  (SEQ / NCHK)         // 64 steps per chunk
#define EPSV  1e-5f

typedef __attribute__((ext_vector_type(16))) __bf16 v16bf;
typedef __attribute__((ext_vector_type(8)))  float  v8f;

union FragAB { v16bf v; uint4 q[2]; };
union BF16x8  { uint4 q;    unsigned short s[8];  };
union BF16x4  { uint2 d;    unsigned short s[4];  };

static __device__ __forceinline__ unsigned short f2bfu(float f) {
  union { float f; unsigned u; } v; v.f = f;
  unsigned u = v.u + 0x7FFFu + ((v.u >> 16) & 1u);   // round-to-nearest-even
  return (unsigned short)(u >> 16);
}
static __device__ __forceinline__ float bfu2f(unsigned short s) {
  union { unsigned u; float f; } v; v.u = ((unsigned)s) << 16;
  return v.f;
}
static __device__ __forceinline__ float siluf(float x) {
  return x / (1.f + __expf(-x));
}

// ---------------------------------------------------------------------------
__global__ void zero_f32(float* p, int n) {
  int i = blockIdx.x * blockDim.x + threadIdx.x;
  if (i < n) p[i] = 0.f;
}

// convert f32 weight [rows,K] -> bf16 [padRows,K], zero-filled pad rows
__global__ void f32_to_bf16_pad(const float* __restrict__ s, unsigned short* __restrict__ d,
                                int rows, int padRows, int K) {
  int i = blockIdx.x * blockDim.x + threadIdx.x;
  if (i >= padRows * K) return;
  int r = i / K;
  d[i] = (r < rows) ? f2bfu(s[i]) : (unsigned short)0;
}

// conv1d(k=3,pad=1) + batchnorm + relu -> u[tok,128] (bf16)
__global__ void __launch_bounds__(256)
conv_bn_relu(const float* __restrict__ x, int in_ch,
             const float* __restrict__ cw, const float* __restrict__ cb,
             const float* __restrict__ g,  const float* __restrict__ bb,
             const float* __restrict__ bm, const float* __restrict__ bv,
             unsigned short* __restrict__ u) {
  int idx = blockIdx.x * blockDim.x + threadIdx.x;       // NTOK * DM
  if (idx >= NTOK * DM) return;
  int c = idx & (DM - 1);
  int t = idx >> 7;                                      // token = b*SEQ + l
  int b = t >> 12, l = t & (SEQ - 1);
  float acc = cb[c];
  for (int ic = 0; ic < in_ch; ++ic)
    #pragma unroll
    for (int k = 0; k < 3; ++k) {
      int ll = l + k - 1;
      if (ll >= 0 && ll < SEQ)
        acc += cw[(c * in_ch + ic) * 3 + k] * x[((size_t)b * in_ch + ic) * SEQ + ll];
    }
  acc = (acc - bm[c]) * rsqrtf(bv[c] + EPSV) * g[c] + bb[c];
  u[(size_t)t * DM + c] = f2bfu(fmaxf(acc, 0.f));
}

// ---------------------------------------------------------------------------
// GEMM: Out[M,16*NT](bf16) = A[M,K](bf16) x W[16*NT,K]^T(bf16), f32 accum.
// One wave owns a 16-token tile; K = 32*KT, N = 16*NT (compile-time, padded).
// A frag (16x32 bf16, ISA 7.12.2): lane = row; per lane two contiguous 16B runs
//   k = 32*kt + 8*half + [0..7]  and  k = 32*kt + 8*half + 16 + [0..7].
// B frag (32x16 bf16): lane = col; same two 16B runs as A
//   k = 32*kt + 8*half + [0..7]  and  k = 32*kt + 8*half + 16 + [0..7].
// Grid sized exactly (M/16 waves); no divergence -> EXEC all 1s at every WMMA.
template<int KT, int NT>
__global__ void __launch_bounds__(256)
gemm_bf16_wmma(const unsigned short* __restrict__ A,
               const unsigned short* __restrict__ W,
               unsigned short* __restrict__ Out) {
  constexpr int K = 32 * KT;
  constexpr int N = 16 * NT;
  const int lane = threadIdx.x & 31;
  const int mt   = (blockIdx.x * blockDim.x + threadIdx.x) >> 5;  // token tile
  const int half = lane >> 4;
  const int l16  = lane & 15;

  FragAB a[KT];
  const unsigned short* Arow = A + (size_t)(mt * 16 + l16) * K + 8 * half;
  #pragma unroll
  for (int kt = 0; kt < KT; ++kt) {
    a[kt].q[0] = *(const uint4*)(Arow + 32 * kt);
    a[kt].q[1] = *(const uint4*)(Arow + 32 * kt + 16);
  }

  const unsigned short* Wrow = W + (size_t)(l16)*K + 8 * half;    // col = l16
  for (int nt = 0; nt < NT; ++nt) {
    const unsigned short* Wp = Wrow + (size_t)nt * 16 * K;
    v8f acc = {};
    #pragma unroll
    for (int kt = 0; kt < KT; ++kt) {
      FragAB bf;
      bf.q[0] = *(const uint4*)(Wp + 32 * kt);
      bf.q[1] = *(const uint4*)(Wp + 32 * kt + 16);
      acc = __builtin_amdgcn_wmma_f32_16x16x32_bf16(false, a[kt].v, false, bf.v,
                                                    (short)0, acc, false, false);
    }
    #pragma unroll
    for (int j = 0; j < 8; ++j) {
      const int mm = mt * 16 + j + 8 * half;           // C/D layout: M = j + 8*half
      Out[(size_t)mm * N + nt * 16 + l16] = f2bfu(acc[j]);
    }
  }
}

// depthwise causal conv (k=4, left pad 3) on x-half of xz, + SiLU -> xc (bf16)
__global__ void __launch_bounds__(256)
dconv_silu(const unsigned short* __restrict__ xz,
           const float* __restrict__ dcw, const float* __restrict__ dcb,
           unsigned short* __restrict__ xc) {
  int idx = blockIdx.x * blockDim.x + threadIdx.x;     // NTOK * DI
  if (idx >= NTOK * DI) return;
  int d = idx & (DI - 1);
  int t = idx >> 8;
  int b = t >> 12, l = t & (SEQ - 1);
  float acc = dcb[d];
  #pragma unroll
  for (int k = 0; k < 4; ++k) {
    int ll = l + k - 3;
    if (ll >= 0)
      acc += dcw[d * 4 + k] * bfu2f(xz[((size_t)b * SEQ + ll) * (2 * DI) + d]);
  }
  xc[(size_t)t * DI + d] = f2bfu(siluf(acc));
}

// delta = softplus(dt @ dt_proj_w^T + b); dt = dbc[:, :8] (one b128 load)
__global__ void __launch_bounds__(256)
dt_softplus(const unsigned short* __restrict__ dbc,
            const float* __restrict__ dtw, const float* __restrict__ dtb,
            unsigned short* __restrict__ delta) {
  int idx = blockIdx.x * blockDim.x + threadIdx.x;     // NTOK * DI
  if (idx >= NTOK * DI) return;
  int d = idx & (DI - 1);
  int t = idx >> 8;
  BF16x8 dt8;
  dt8.q = *(const uint4*)(dbc + (size_t)t * DBCP);
  float acc = dtb[d];
  #pragma unroll
  for (int r = 0; r < DR; ++r)
    acc += dtw[d * DR + r] * bfu2f(dt8.s[r]);
  float sp = (acc > 20.f) ? acc : log1pf(__expf(acc));
  delta[(size_t)t * DI + d] = f2bfu(sp);
}

// --- chunked selective scan: h_t = exp(delta*A)*h + delta*x*B (linear in h) ---
// Pass A: per-chunk summary (prod of dA, affine accumulator) -- 262144 threads
__global__ void __launch_bounds__(256)
scan_partial(const unsigned short* __restrict__ xc,
             const unsigned short* __restrict__ delta,
             const unsigned short* __restrict__ dbc,
             const float* __restrict__ A_log,
             float* __restrict__ Ap, float* __restrict__ Bc) {
  int id = blockIdx.x * blockDim.x + threadIdx.x;      // BATCH*DI*NCHK
  if (id >= BATCH * DI * NCHK) return;
  int c = id % NCHK, d = (id / NCHK) % DI, b = id / (NCHK * DI);
  float An[DSN], ap[DSN], bc[DSN];
  #pragma unroll
  for (int n = 0; n < DSN; ++n) { An[n] = -__expf(A_log[d * DSN + n]); ap[n] = 1.f; bc[n] = 0.f; }
  for (int s = 0; s < CLEN; ++s) {
    size_t t = (size_t)b * SEQ + c * CLEN + s;
    float xv = bfu2f(xc[t * DI + d]);
    float dv = bfu2f(delta[t * DI + d]);
    float du = dv * xv;
    union { uint4 q[2]; unsigned short s16[16]; } Bv;  // cols 8..23 (aligned b128 x2)
    Bv.q[0] = *(const uint4*)(dbc + t * DBCP + 8);
    Bv.q[1] = *(const uint4*)(dbc + t * DBCP + 16);
    #pragma unroll
    for (int n = 0; n < DSN; ++n) {
      float dA = __expf(dv * An[n]);
      ap[n] *= dA;
      bc[n] = dA * bc[n] + du * bfu2f(Bv.s16[n]);
    }
  }
  size_t o = ((size_t)(b * DI + d) * NCHK + c) * DSN;
  #pragma unroll
  for (int n = 0; n < DSN; ++n) { Ap[o + n] = ap[n]; Bc[o + n] = bc[n]; }
}

// Pass B: sequential combine of 64 chunk summaries per (b,d) -> chunk-entry states
__global__ void scan_combine(const float* __restrict__ Ap, const float* __restrict__ Bc,
                             float* __restrict__ H0) {
  int id = blockIdx.x * blockDim.x + threadIdx.x;      // BATCH*DI
  if (id >= BATCH * DI) return;
  float h[DSN];
  #pragma unroll
  for (int n = 0; n < DSN; ++n) h[n] = 0.f;
  for (int c = 0; c < NCHK; ++c) {
    size_t o = ((size_t)id * NCHK + c) * DSN;
    #pragma unroll
    for (int n = 0; n < DSN; ++n) {
      H0[o + n] = h[n];
      h[n] = Ap[o + n] * h[n] + Bc[o + n];
    }
  }
}

// Pass C: re-scan each chunk from its true entry state; fuse y = h.C + x*Dp,
// gate with silu(z) -> yg (bf16)
__global__ void __launch_bounds__(256)
scan_final(const unsigned short* __restrict__ xc,
           const unsigned short* __restrict__ delta,
           const unsigned short* __restrict__ dbc,
           const unsigned short* __restrict__ xz,
           const float* __restrict__ A_log, const float* __restrict__ Dp,
           const float* __restrict__ H0,
           unsigned short* __restrict__ yg) {
  int id = blockIdx.x * blockDim.x + threadIdx.x;      // BATCH*DI*NCHK
  if (id >= BATCH * DI * NCHK) return;
  int c = id % NCHK, d = (id / NCHK) % DI, b = id / (NCHK * DI);
  float An[DSN], h[DSN];
  size_t o = ((size_t)(b * DI + d) * NCHK + c) * DSN;
  #pragma unroll
  for (int n = 0; n < DSN; ++n) { An[n] = -__expf(A_log[d * DSN + n]); h[n] = H0[o + n]; }
  float DpV = Dp[d];
  for (int s = 0; s < CLEN; ++s) {
    size_t t = (size_t)b * SEQ + c * CLEN + s;
    float xv = bfu2f(xc[t * DI + d]);
    float dv = bfu2f(delta[t * DI + d]);
    float du = dv * xv;
    union { uint4 q[2]; unsigned short s16[16]; } Bv, Cv;
    Bv.q[0] = *(const uint4*)(dbc + t * DBCP + 8);     // B: cols 8..23
    Bv.q[1] = *(const uint4*)(dbc + t * DBCP + 16);
    Cv.q[0] = *(const uint4*)(dbc + t * DBCP + 24);    // C: cols 24..39
    Cv.q[1] = *(const uint4*)(dbc + t * DBCP + 32);
    float y = 0.f;
    #pragma unroll
    for (int n = 0; n < DSN; ++n) {
      float dA = __expf(dv * An[n]);
      h[n] = dA * h[n] + du * bfu2f(Bv.s16[n]);
      y += h[n] * bfu2f(Cv.s16[n]);
    }
    float zv = bfu2f(xz[t * (2 * DI) + DI + d]);
    yg[t * DI + d] = f2bfu((y + xv * DpV) * siluf(zv));
  }
}

// LayerNorm over 128 channels (one wave32 per token, 4 contiguous ch per lane)
// + mean-pool over L via f32 global atomics into the fused output region.
__global__ void __launch_bounds__(256)
ln_pool(const unsigned short* __restrict__ om,
        const float* __restrict__ g, const float* __restrict__ bta,
        float* __restrict__ fused, int brOff) {
  int wave = (blockIdx.x * blockDim.x + threadIdx.x) >> 5;   // token id
  int lane = threadIdx.x & 31;
  if (wave >= NTOK) return;
  int b = wave >> 12;
  BF16x4 r4;
  r4.d = *(const uint2*)(om + (size_t)wave * DM + lane * 4); // b64 load
  float v[4]; float s = 0.f;
  #pragma unroll
  for (int j = 0; j < 4; ++j) { v[j] = bfu2f(r4.s[j]); s += v[j]; }
  #pragma unroll
  for (int off = 16; off > 0; off >>= 1) s += __shfl_xor(s, off, 32);
  float mu = s * (1.f / DM);
  float sv = 0.f;
  #pragma unroll
  for (int j = 0; j < 4; ++j) { float dd = v[j] - mu; sv += dd * dd; }
  #pragma unroll
  for (int off = 16; off > 0; off >>= 1) sv += __shfl_xor(sv, off, 32);
  float rs = rsqrtf(sv * (1.f / DM) + EPSV);
  #pragma unroll
  for (int j = 0; j < 4; ++j) {
    int ch = lane * 4 + j;
    float val = (v[j] - mu) * rs * g[ch] + bta[ch];
    atomicAdd(&fused[(size_t)b * (2 * DM) + brOff + ch], val * (1.f / SEQ));
  }
}

// fused[16,256] -> relu(w1) -> w2 -> logits[16,10]
__global__ void classifier(const float* __restrict__ fused,
                           const float* __restrict__ w1, const float* __restrict__ b1,
                           const float* __restrict__ w2, const float* __restrict__ b2,
                           float* __restrict__ out) {
  __shared__ float h[64];
  int b = blockIdx.x, j = threadIdx.x;
  float acc = b1[j];
  for (int i = 0; i < 2 * DM; ++i) acc += fused[b * (2 * DM) + i] * w1[j * (2 * DM) + i];
  h[j] = fmaxf(acc, 0.f);
  __syncthreads();
  if (j < 10) {
    float o = b2[j];
    #pragma unroll
    for (int i = 0; i < 64; ++i) o += h[i] * w2[j * 64 + i];
    out[b * 10 + j] = o;
  }
}

// ---------------------------------------------------------------------------
extern "C" void kernel_launch(void* const* d_in, const int* in_sizes, int n_in,
                              void* d_out, int out_size, void* d_ws, size_t ws_size,
                              hipStream_t stream) {
  (void)in_sizes; (void)n_in; (void)out_size; (void)ws_size;

  const float* time_data = (const float*)d_in[0];
  const float* freq_data = (const float*)d_in[1];
  // per-branch params, insertion order of _branch_params
  const float* P[2][17];
  for (int br = 0; br < 2; ++br)
    for (int i = 0; i < 17; ++i) P[br][i] = (const float*)d_in[2 + br * 17 + i];
  enum { CONV_W, CONV_B, BN_G, BN_B, BN_M, BN_V, IN_PROJ_W, DCONV_W, DCONV_B,
         X_PROJ_W, DT_PROJ_W, DT_PROJ_B, A_LOG, DP, OUT_PROJ_W, LN_G, LN_B };
  const float* cls_w1 = (const float*)d_in[36];
  const float* cls_b1 = (const float*)d_in[37];
  const float* cls_w2 = (const float*)d_in[38];
  const float* cls_b2 = (const float*)d_in[39];

  // workspace layout (256B aligned)
  char* ws = (char*)d_ws;
  size_t off = 0;
  auto alloc = [&](size_t bytes) { size_t o = off; off += (bytes + 255) & ~(size_t)255; return o; };
  unsigned short* u_bf    = (unsigned short*)(ws + alloc((size_t)NTOK * DM * 2));
  unsigned short* xz_bf   = (unsigned short*)(ws + alloc((size_t)NTOK * 2 * DI * 2));
  unsigned short* xc_bf   = (unsigned short*)(ws + alloc((size_t)NTOK * DI * 2));
  unsigned short* dbc_bf  = (unsigned short*)(ws + alloc((size_t)NTOK * DBCP * 2));
  unsigned short* delta_bf= (unsigned short*)(ws + alloc((size_t)NTOK * DI * 2));
  unsigned short* yg_bf   = (unsigned short*)(ws + alloc((size_t)NTOK * DI * 2));
  unsigned short* om_bf   = (unsigned short*)(ws + alloc((size_t)NTOK * DM * 2));
  unsigned short* wb_in   = (unsigned short*)(ws + alloc((size_t)2 * DI * DM * 2));
  unsigned short* wb_x    = (unsigned short*)(ws + alloc((size_t)DBCP * DI * 2));
  unsigned short* wb_out  = (unsigned short*)(ws + alloc((size_t)DM * DI * 2));
  float* scanAp = (float*)(ws + alloc((size_t)BATCH * DI * NCHK * DSN * 4));
  float* scanBc = (float*)(ws + alloc((size_t)BATCH * DI * NCHK * DSN * 4));
  float* scanH0 = (float*)(ws + alloc((size_t)BATCH * DI * NCHK * DSN * 4));

  float* logits = (float*)d_out;                 // [16,10]
  float* fused  = (float*)d_out + BATCH * 10;    // [16,256]

  const int nOut = BATCH * 10 + BATCH * 2 * DM;
  zero_f32<<<(nOut + 255) / 256, 256, 0, stream>>>((float*)d_out, nOut);

  const int GW = (NTOK / 16) * 32 / 256;         // 512 blocks: exact wave coverage

  for (int br = 0; br < 2; ++br) {
    const float* data = br ? freq_data : time_data;
    const int in_ch = br ? 2 : 1;
    const float* const* p = P[br];

    f32_to_bf16_pad<<<(2 * DI * DM + 255) / 256, 256, 0, stream>>>(
        p[IN_PROJ_W], wb_in, 2 * DI, 2 * DI, DM);
    f32_to_bf16_pad<<<(DBCP * DI + 255) / 256, 256, 0, stream>>>(
        p[X_PROJ_W], wb_x, DBCW, DBCP, DI);                 // zero-pad rows 40..47
    f32_to_bf16_pad<<<(DM * DI + 255) / 256, 256, 0, stream>>>(
        p[OUT_PROJ_W], wb_out, DM, DM, DI);

    conv_bn_relu<<<NTOK * DM / 256, 256, 0, stream>>>(
        data, in_ch, p[CONV_W], p[CONV_B], p[BN_G], p[BN_B], p[BN_M], p[BN_V], u_bf);

    gemm_bf16_wmma<4, 32><<<GW, 256, 0, stream>>>(u_bf,  wb_in,  xz_bf);   // K=128,N=512
    dconv_silu<<<NTOK * DI / 256, 256, 0, stream>>>(xz_bf, p[DCONV_W], p[DCONV_B], xc_bf);
    gemm_bf16_wmma<8, 3><<<GW, 256, 0, stream>>>(xc_bf, wb_x,  dbc_bf);    // K=256,N=48
    dt_softplus<<<NTOK * DI / 256, 256, 0, stream>>>(dbc_bf, p[DT_PROJ_W], p[DT_PROJ_B], delta_bf);

    scan_partial<<<BATCH * DI * NCHK / 256, 256, 0, stream>>>(
        xc_bf, delta_bf, dbc_bf, p[A_LOG], scanAp, scanBc);
    scan_combine<<<(BATCH * DI + 255) / 256, 256, 0, stream>>>(scanAp, scanBc, scanH0);
    scan_final<<<BATCH * DI * NCHK / 256, 256, 0, stream>>>(
        xc_bf, delta_bf, dbc_bf, xz_bf, p[A_LOG], p[DP], scanH0, yg_bf);

    gemm_bf16_wmma<8, 8><<<GW, 256, 0, stream>>>(yg_bf, wb_out, om_bf);    // K=256,N=128
    ln_pool<<<NTOK * 32 / 256, 256, 0, stream>>>(om_bf, p[LN_G], p[LN_B], fused, br * DM);
  }

  classifier<<<BATCH, 64, 0, stream>>>(fused, cls_w1, cls_b1, cls_w2, cls_b2, logits);
}
